// TPGGMLLinear_4707284156754
// MI455X (gfx1250) — compile-verified, matrix operand order checked by reading, not run on hardware
//
#include <hip/hip_runtime.h>
#include <hip/hip_fp16.h>
#include <stdint.h>

// ---- problem constants (from reference) ----
#define IN_F   8192
#define OUT_F  8192
#define BATCH  16
#define KB_TOT (IN_F / 32)      // 256 Q8_0 blocks per output row
#define BLK_BYTES 34            // 2B fp16 scale + 32 int8 quants
#define NSPLIT 4
#define KB_PER (KB_TOT / NSPLIT) // 64 k-blocks per wave

// workspace layout:
//   [0, 256KB)            : A fragments, f16, [256 ksteps][32 lanes][16 halfs]
//   [256KB, 256KB+2MB)    : f32 partials [4 splits][16][8192]
#define AFRAG_BYTES ((size_t)KB_TOT * 32 * 16 * 2)          // 262144
#define PART_ELEMS  ((size_t)NSPLIT * BATCH * OUT_F)        // 524288 f32

typedef __attribute__((ext_vector_type(16))) _Float16 v16h;
typedef __attribute__((ext_vector_type(8)))  float    v8f;

// ---------------------------------------------------------------------------
// Prep: pack x (f32) into the exact wave32 16-bit A-fragment layout.
// Lane L (0-15): M=L,    K = ks*32 + {0..7, 16..23}
// Lane L (16-31): M=L-16, K = ks*32 + {8..15, 24..31}
// Stored contiguously: afrag[(ks*32 + lane)*16 + j]  (32B per lane -> b128 x2)
// ---------------------------------------------------------------------------
__global__ void __launch_bounds__(32)
q8_prep_afrag(const float* __restrict__ x, _Float16* __restrict__ afrag) {
    const int ks   = blockIdx.x;          // 0..255
    const int lane = threadIdx.x;         // 0..31
    const int m    = lane & 15;
    const int h    = lane >> 4;
    const float* xp = x + (size_t)m * IN_F + (size_t)ks * 32 + h * 8;
    _Float16* op = afrag + ((size_t)ks * 32 + lane) * 16;
#pragma unroll
    for (int j = 0; j < 8; ++j) op[j]     = (_Float16)xp[j];
#pragma unroll
    for (int j = 0; j < 8; ++j) op[8 + j] = (_Float16)xp[16 + j];
}

// fp16 magic dequant helper: h2 = d * q  via  fma(1152+q, d, -1152*d)
static __device__ __forceinline__ uint32_t h2_fma(uint32_t magic, __half2 d2, __half2 off2) {
    union { uint32_t u; __half2 h; } a, r;
    a.u = magic;
    r.h = __hfma2(a.h, d2, off2);
    return r.u;
}

// ---------------------------------------------------------------------------
// Main GEMM: one wave per (16x16 N-tile, k-split). 512 tiles x 4 splits.
// Streams raw Q8_0 bytes once; dequant in regs; v_wmma_f32_16x16x32_f16.
// ---------------------------------------------------------------------------
__global__ void __launch_bounds__(128)
q8_wmma_gemm(const uint8_t* __restrict__ qw,
             const _Float16* __restrict__ afrag,
             float* __restrict__ partials) {
    const int lane  = threadIdx.x & 31;
    const int split = threadIdx.x >> 5;            // 0..3
    const int n0    = blockIdx.x * 16;             // output column tile
    const int col   = n0 + (lane & 15);            // this lane's output column
    const int h     = lane >> 4;                   // 0: K 0-7/16-23, 1: K 8-15/24-31

    const int kb0 = split * KB_PER;
    const uint8_t* blk = qw + ((size_t)col * KB_TOT + kb0) * BLK_BYTES;
    const _Float16* ap = afrag + ((size_t)kb0 * 32 + lane) * 16;

    const __half2 neg1152 = __floats2half2_rn(-1152.0f, -1152.0f);
    v8f acc = {0.f, 0.f, 0.f, 0.f, 0.f, 0.f, 0.f, 0.f};

#pragma unroll 2
    for (int kb = 0; kb < KB_PER; ++kb) {
        // per-block fp16 scale (2B aligned)
        union { uint16_t u; __half hh; } dr;
        dr.u = *(const uint16_t*)blk;
        const __half2 d2   = __half2half2(dr.hh);
        const __half2 off2 = __hmul2(d2, neg1152);

        // 16 int8 quants for this lane (2B aligned only -> u16 loads)
        const uint16_t* qa = (const uint16_t*)(blk + 2 + h * 8);    // K lo chunk
        const uint16_t* qb = (const uint16_t*)(blk + 18 + h * 8);   // K hi chunk
        uint32_t w0 = (uint32_t)qa[0] | ((uint32_t)qa[1] << 16);
        uint32_t w1 = (uint32_t)qa[2] | ((uint32_t)qa[3] << 16);
        uint32_t w2 = (uint32_t)qb[0] | ((uint32_t)qb[1] << 16);
        uint32_t w3 = (uint32_t)qb[2] | ((uint32_t)qb[3] << 16);

        // A fragment: 32B contiguous, coalesced, hot in L2
        v16h a = *(const v16h*)ap;

        // Build B fragment: (q^0x80)|0x6400 == fp16(1152+q); scale via pk_fma
        union { v16h v; uint32_t u[8]; } b;
        uint32_t words[4] = { w0, w1, w2, w3 };
#pragma unroll
        for (int i = 0; i < 4; ++i) {
            uint32_t xw = words[i] ^ 0x80808080u;
            uint32_t lo = __builtin_amdgcn_perm(0x64646464u, xw, 0x04010400u); // {64,b1,64,b0}
            uint32_t hi = __builtin_amdgcn_perm(0x64646464u, xw, 0x04030402u); // {64,b3,64,b2}
            b.u[2 * i]     = h2_fma(lo, d2, off2);
            b.u[2 * i + 1] = h2_fma(hi, d2, off2);
        }

        acc = __builtin_amdgcn_wmma_f32_16x16x32_f16(
            /*neg_a=*/false, a, /*neg_b=*/false, b.v,
            /*c_mod=*/(short)0, acc, /*reuse_a=*/false, /*reuse_b=*/false);

        blk += BLK_BYTES;
        ap  += 32 * 16;
    }

    // C/D layout: lanes 0-15 -> M=r, lanes 16-31 -> M=8+r; N = col
    float* pout = partials + (size_t)split * BATCH * OUT_F;
#pragma unroll
    for (int r = 0; r < 8; ++r)
        pout[(size_t)(r + h * 8) * OUT_F + col] = acc[r];
}

// ---------------------------------------------------------------------------
// Deterministic reduction: out = sum of 4 partials + bias (writes all of d_out)
// ---------------------------------------------------------------------------
__global__ void __launch_bounds__(256)
q8_reduce(const float* __restrict__ partials, const float* __restrict__ bias,
          float* __restrict__ out) {
    const int idx = blockIdx.x * blockDim.x + threadIdx.x; // 0..131071
    const int n = idx & (OUT_F - 1);
    float s = bias[n];
#pragma unroll
    for (int sp = 0; sp < NSPLIT; ++sp)
        s += partials[(size_t)sp * BATCH * OUT_F + idx];
    out[idx] = s;
}

extern "C" void kernel_launch(void* const* d_in, const int* in_sizes, int n_in,
                              void* d_out, int out_size, void* d_ws, size_t ws_size,
                              hipStream_t stream) {
    (void)in_sizes; (void)n_in; (void)out_size; (void)ws_size;
    const float*   x    = (const float*)d_in[0];
    const uint8_t* qw   = (const uint8_t*)d_in[1];
    const float*   bias = (const float*)d_in[2];
    float* out = (float*)d_out;

    _Float16* afrag    = (_Float16*)d_ws;                          // 256 KB
    float*    partials = (float*)((uint8_t*)d_ws + AFRAG_BYTES);   // 2 MB

    // 1) pack x into A fragments (one wave per k-step)
    q8_prep_afrag<<<KB_TOT, 32, 0, stream>>>(x, afrag);
    // 2) fused dequant + WMMA GEMM: 512 N-tiles x 4 k-splits (4 waves/block)
    q8_wmma_gemm<<<OUT_F / 16, 128, 0, stream>>>(qw, afrag, partials);
    // 3) deterministic k-split reduction + bias
    q8_reduce<<<(BATCH * OUT_F) / 256, 256, 0, stream>>>(partials, bias, out);
}
